// CrossHybridAttention_83837761618244
// MI455X (gfx1250) — matrix operand order repaired and checked by reading, not compiled, not run on hardware
//
#include <hip/hip_runtime.h>

// CDNA5 / gfx1250, wave32. WMMA f32_16x16x32_f16 for both projection GEMM and
// attention-score GEMM (K=32 == DH, so one WMMA per head per 16-key tile).
//
// Q/K are stored by the projection kernels directly in WMMA *fragment order*:
// one contiguous 1 KB record per (b, h, 16-row tile), element order = the ISA's
// per-lane A-/B-matrix VGPR layout. The attention kernel's fragment loads are
// then frag_base + lane*32B: perfectly coalesced 1 KB wave transactions.
//
// Attention keeps scores + head-mean accumulator fully in VGPRs; softmax stats
// reduce via shfl_xor within 16-lane halves + a 1 KB LDS cross-wave combine.

typedef __attribute__((ext_vector_type(16))) _Float16 v16h;
typedef __attribute__((ext_vector_type(8)))  float    v8f;

#define B_  8
#define C_  128
#define LX_ 4096
#define LY_ 1024
#define H_  4
#define DH_ 32
#define FRAG_HALVES 512   // 16 rows x 32 d, one WMMA operand tile

union AB16 { v16h v; uint4 q[2]; };

// -----------------------------------------------------------------------------
// Projection: frag[(b*H + h)*(L/16) + tile] <- f16( x^T @ W^T + bias )
//   xsrc: (B, C, L) f32, W: (C, C) row-major (o, c) f32, bias: (C) f32
// Block = 256 threads (8 waves), handles 16 rows (l) x 128 cols (o = h*32+d).
// Each wave owns one 16-wide o-tile; 4 WMMAs accumulate over K = C = 128.
// MODE 0: store in A-fragment order (Q). MODE 1: store in B-fragment order (K).
// -----------------------------------------------------------------------------
template <int MODE>
__global__ void proj_kernel(const float* __restrict__ xsrc,
                            const float* __restrict__ W,
                            const float* __restrict__ bias,
                            _Float16* __restrict__ outp,
                            int L)
{
    __shared__ __align__(16) _Float16 w_lds[C_ * C_];   // 32 KB, f16 weights
    __shared__ __align__(16) _Float16 x_lds[16 * C_];   // 4 KB, transposed x tile

    const int tid  = threadIdx.x;
    const int b    = blockIdx.y;
    const int tile = blockIdx.x;
    const int l0   = tile * 16;

    for (int i = tid; i < C_ * C_; i += 256)
        w_lds[i] = (_Float16)W[i];

    // x_lds[l][c] = x[b][c][l0+l]; consecutive i -> coalesced l reads.
    for (int i = tid; i < 16 * C_; i += 256) {
        int c = i >> 4, l = i & 15;
        x_lds[l * C_ + c] = (_Float16)xsrc[(size_t)b * C_ * L + (size_t)c * L + l0 + l];
    }
    __syncthreads();

    const int wave = tid >> 5;
    const int lane = tid & 31;
    const int n    = lane & 15;
    const int hi   = lane >> 4;
    const int o0   = wave * 16;

    v8f acc = {};
    #pragma unroll
    for (int kc = 0; kc < 4; ++kc) {
        const int c0 = kc * 32;
        AB16 a, bm;
        // A 16x32 f16: lane row m = lane&15; halves 0-7 = K(hi*8..), 8-15 = K(16+hi*8..)
        a.q[0] = *(const uint4*)&x_lds[n * C_ + c0 + hi * 8];
        a.q[1] = *(const uint4*)&x_lds[n * C_ + c0 + 16 + hi * 8];
        // B 32x16 f16: lane column n; halves 0-7 = K(hi*8..), 8-15 = K(16+hi*8..)
        bm.q[0] = *(const uint4*)&w_lds[(o0 + n) * C_ + c0 + hi * 8];
        bm.q[1] = *(const uint4*)&w_lds[(o0 + n) * C_ + c0 + 16 + hi * 8];
        acc = __builtin_amdgcn_wmma_f32_16x16x32_f16(
                  false, a.v, false, bm.v, (short)0, acc, false, false);
    }

    // This lane's output column: channel o = o0+n -> head h, within-head d.
    const int o = o0 + n;
    const int h = o >> 5;          // DH_ == 32
    const int d = o & 31;
    const float bo = bias[o];

    _Float16* frag = outp + ((size_t)(b * H_ + h) * (L / 16) + tile) * FRAG_HALVES;

    #pragma unroll
    for (int r = 0; r < 8; ++r) {
        const int m = hi * 8 + r;                      // row within the 16-row tile
        int lidx, half;
        if (MODE == 0) {                               // A-fragment order (Q)
            lidx = (((d >> 3) & 1) << 4) | m;          // halves 0-7: K(hiA*8..), 8-15: K(16+hiA*8..)
            half = (d & 7) | ((d >> 4) << 3);
        } else {                                       // B-fragment order (K)
            lidx = (((d >> 3) & 1) << 4) | m;          // halves 0-7: K(hiB*8..), 8-15: K(16+hiB*8..)
            half = (d & 7) | ((d >> 4) << 3);
        }
        frag[lidx * 16 + half] = (_Float16)(acc[r] + bo);
    }
}

// -----------------------------------------------------------------------------
// Attention: block = (b, 16 query rows), 256 threads = 8 waves.
// Wave w owns key tiles kt = w + 8*i, i = 0..7 (same tiles every head), so both
// the per-head scores s[8] and the mean-over-heads accumulator acc[8] live in
// VGPRs (lane element = D[row hi*8+r][key (w+8i)*16 + n]).
// Fragment loads: contiguous, frag_base + lane*32B (coalesced 1 KB per wave).
// Softmax stats: in-register reduce over i, shfl_xor over the 16 key columns,
// 8x16 LDS arrays for the cross-wave combine.
// -----------------------------------------------------------------------------
__global__ void attn_kernel(const _Float16* __restrict__ Qf,  // A-frag records
                            const _Float16* __restrict__ Kf,  // B-frag records
                            float* __restrict__ cross)        // (B, LX)
{
    __shared__ float pmax[8][16];   // [wave][row] partial max
    __shared__ float psum[8][16];   // [wave][row] partial sum

    const int tid  = threadIdx.x;
    const int b    = blockIdx.y;
    const int qt   = blockIdx.x;            // query tile index
    const int q0   = qt * 16;
    const int wave = tid >> 5;
    const int lane = tid & 31;
    const int n    = lane & 15;
    const int hi   = lane >> 4;

    const float scale = 0.17677669529663687f;   // 1/sqrt(DH)

    v8f acc[8];
    #pragma unroll
    for (int i = 0; i < 8; ++i) acc[i] = (v8f){};

    for (int h = 0; h < H_; ++h) {
        // A fragment (Q tile 16x32 f16): one coalesced 32B read per lane.
        AB16 a;
        const _Float16* qp =
            Qf + ((size_t)(b * H_ + h) * (LX_ / 16) + qt) * FRAG_HALVES + lane * 16;
        a.q[0] = *(const uint4*)(qp);
        a.q[1] = *(const uint4*)(qp + 8);

        // 8 key tiles per wave, fully unrolled: loads hoisted ahead of WMMAs.
        v8f s[8];
        #pragma unroll
        for (int i = 0; i < 8; ++i) {
            const int kt = wave + i * 8;
            const _Float16* kp =
                Kf + ((size_t)(b * H_ + h) * (LY_ / 16) + kt) * FRAG_HALVES + lane * 16;
            AB16 bm;
            bm.q[0] = *(const uint4*)(kp);
            bm.q[1] = *(const uint4*)(kp + 8);
            v8f d = {};
            s[i] = __builtin_amdgcn_wmma_f32_16x16x32_f16(
                       false, a.v, false, bm.v, (short)0, d, false, false);
        }
        #pragma unroll
        for (int i = 0; i < 8; ++i)
            #pragma unroll
            for (int r = 0; r < 8; ++r) s[i][r] *= scale;

        // ---- row max: over i in-register, over n via shfl, over waves via LDS.
        float pm[8];
        #pragma unroll
        for (int r = 0; r < 8; ++r) {
            float m = s[0][r];
            #pragma unroll
            for (int i = 1; i < 8; ++i) m = fmaxf(m, s[i][r]);
            #pragma unroll
            for (int msk = 1; msk < 16; msk <<= 1) m = fmaxf(m, __shfl_xor(m, msk, 32));
            pm[r] = m;
        }
        if (n == 0) {
            #pragma unroll
            for (int r = 0; r < 8; ++r) pmax[wave][hi * 8 + r] = pm[r];
        }
        __syncthreads();

        float rmax[8];
        #pragma unroll
        for (int r = 0; r < 8; ++r) {
            float m = pmax[0][hi * 8 + r];
            #pragma unroll
            for (int w = 1; w < 8; ++w) m = fmaxf(m, pmax[w][hi * 8 + r]);
            rmax[r] = m;
        }

        // ---- exp (once) + row sum.
        float ps[8];
        #pragma unroll
        for (int r = 0; r < 8; ++r) ps[r] = 0.0f;
        #pragma unroll
        for (int i = 0; i < 8; ++i)
            #pragma unroll
            for (int r = 0; r < 8; ++r) {
                const float e = __expf(s[i][r] - rmax[r]);
                s[i][r] = e;
                ps[r] += e;
            }
        #pragma unroll
        for (int r = 0; r < 8; ++r)
            #pragma unroll
            for (int msk = 1; msk < 16; msk <<= 1) ps[r] += __shfl_xor(ps[r], msk, 32);
        if (n == 0) {
            #pragma unroll
            for (int r = 0; r < 8; ++r) psum[wave][hi * 8 + r] = ps[r];
        }
        __syncthreads();

        #pragma unroll
        for (int r = 0; r < 8; ++r) {
            float ssum = psum[0][hi * 8 + r];
            #pragma unroll
            for (int w = 1; w < 8; ++w) ssum += psum[w][hi * 8 + r];
            const float rinv = 1.0f / ssum;
            #pragma unroll
            for (int i = 0; i < 8; ++i) acc[i][r] = fmaf(s[i][r], rinv, acc[i][r]);
        }
        __syncthreads();   // pmax/psum reused next head
    }

    // ---- cross = max over keys of mean-over-heads probability.
    float pm[8];
    #pragma unroll
    for (int r = 0; r < 8; ++r) {
        float m = acc[0][r];
        #pragma unroll
        for (int i = 1; i < 8; ++i) m = fmaxf(m, acc[i][r]);
        #pragma unroll
        for (int msk = 1; msk < 16; msk <<= 1) m = fmaxf(m, __shfl_xor(m, msk, 32));
        pm[r] = m;
    }
    if (n == 0) {
        #pragma unroll
        for (int r = 0; r < 8; ++r) pmax[wave][hi * 8 + r] = pm[r];
    }
    __syncthreads();
    if (tid < 16) {
        float m = pmax[0][tid];
        #pragma unroll
        for (int w = 1; w < 8; ++w) m = fmaxf(m, pmax[w][tid]);
        cross[(size_t)b * LX_ + q0 + tid] = m * 0.25f;
    }
}

// -----------------------------------------------------------------------------
// Local path + gate blend. One thread per output position l (coalesced x reads).
// -----------------------------------------------------------------------------
__global__ void local_combine_kernel(const float* __restrict__ x,      // (B,C,LX)
                                     const float* __restrict__ w1,     // (32,C)
                                     const float* __restrict__ b1,     // (32)
                                     const float* __restrict__ w2,     // (1,32)
                                     const float* __restrict__ b2,     // (1)
                                     const float* __restrict__ gw,     // (1,2)
                                     const float* __restrict__ gb,     // (1)
                                     const float* __restrict__ cross,  // (B,LX)
                                     float* __restrict__ out)          // (B,LX)
{
    __shared__ float w1s[32 * C_];
    __shared__ float w2s[32];
    __shared__ float b1s[32];

    const int tid = threadIdx.x;
    const int b   = blockIdx.y;
    const int l   = blockIdx.x * 256 + tid;

    for (int i = tid; i < 32 * C_; i += 256) w1s[i] = w1[i];
    if (tid < 32) { w2s[tid] = w2[tid]; b1s[tid] = b1[tid]; }
    __syncthreads();

    float hacc[32];
    #pragma unroll
    for (int o = 0; o < 32; ++o) hacc[o] = b1s[o];

    const float* xp = x + (size_t)b * C_ * LX_ + l;
    for (int c = 0; c < C_; ++c) {
        const float xv = xp[(size_t)c * LX_];
        #pragma unroll
        for (int o = 0; o < 32; ++o)
            hacc[o] = fmaf(w1s[o * C_ + c], xv, hacc[o]);
    }

    float s2 = b2[0];
    #pragma unroll
    for (int o = 0; o < 32; ++o) s2 += w2s[o] * fmaxf(hacc[o], 0.0f);
    const float ls = 1.0f / (1.0f + __expf(-s2));

    const float cs = cross[(size_t)b * LX_ + l];
    const float g  = gw[0] * cs + gw[1] * ls + gb[0];
    const float alpha = 1.0f / (1.0f + __expf(-g));
    out[(size_t)b * LX_ + l] = alpha * cs + (1.0f - alpha) * ls;
}

// -----------------------------------------------------------------------------
extern "C" void kernel_launch(void* const* d_in, const int* in_sizes, int n_in,
                              void* d_out, int out_size, void* d_ws, size_t ws_size,
                              hipStream_t stream)
{
    (void)in_sizes; (void)n_in; (void)out_size; (void)ws_size;

    const float* x   = (const float*)d_in[0];
    const float* y   = (const float*)d_in[1];
    const float* wqk = (const float*)d_in[2];   // in_proj_w (3C, C)
    const float* bqk = (const float*)d_in[3];   // in_proj_b (3C)
    const float* c1w = (const float*)d_in[4];
    const float* c1b = (const float*)d_in[5];
    const float* c2w = (const float*)d_in[6];
    const float* c2b = (const float*)d_in[7];
    const float* gw  = (const float*)d_in[8];
    const float* gb  = (const float*)d_in[9];
    float* out = (float*)d_out;

    char* ws = (char*)d_ws;
    _Float16* Qf   = (_Float16*)ws;                                     // B*H*(LX/16)*512 f16 = 8 MB
    _Float16* Kf   = (_Float16*)(ws + (size_t)B_ * LX_ * C_ * 2);       // B*H*(LY/16)*512 f16 = 2 MB
    float*    crsp = (float*)(ws + (size_t)B_ * LX_ * C_ * 2
                                 + (size_t)B_ * LY_ * C_ * 2);          // B*LX f32

    proj_kernel<0><<<dim3(LX_ / 16, B_), 256, 0, stream>>>(x, wqk,         bqk,      Qf, LX_);
    proj_kernel<1><<<dim3(LY_ / 16, B_), 256, 0, stream>>>(y, wqk + C_*C_, bqk + C_, Kf, LY_);

    attn_kernel<<<dim3(LX_ / 16, B_), 256, 0, stream>>>(Qf, Kf, crsp);

    local_combine_kernel<<<dim3(LX_ / 256, B_), 256, 0, stream>>>(
        x, c1w, c1b, c2w, c2b, gw, gb, crsp, out);
}
